// SparseConvEncoder_20633022890309
// MI455X (gfx1250) — compile-verified, matrix operand order checked by reading, not run on hardware
//
#include <hip/hip_runtime.h>

typedef __attribute__((ext_vector_type(16))) __bf16 v16bf;
typedef __attribute__((ext_vector_type(8)))  float  v8f;

#define CONV_WAVES 4

// ---------------------------------------------------------------- utilities
__global__ __launch_bounds__(256) void fill_i32_kern(int* __restrict__ p, int n, int v) {
  int e = blockIdx.x * blockDim.x + threadIdx.x;
  if (e < n) p[e] = v;
}

__global__ __launch_bounds__(256) void fill_f32_kern(float* __restrict__ p, int n, float v) {
  int e = blockIdx.x * blockDim.x + threadIdx.x;
  if (e < n) p[e] = v;
}

// Invert the padded rulebook: for each valid (k,j), inv[k*P + O[k,j]] = I[k,j].
// O values are distinct per k, so there are no write conflicts.
__global__ __launch_bounds__(256) void invert_rulebook_kern(
    const int* __restrict__ I, const int* __restrict__ O,
    const unsigned char* __restrict__ M, int total, int P, int* __restrict__ inv) {
  int e = blockIdx.x * blockDim.x + threadIdx.x;
  if (e >= total) return;
  if (M[e]) {
    int k = e / P;
    inv[k * P + O[e]] = I[e];
  }
}

// ------------------------------------------------- output-stationary WMMA conv
// One wave owns one 16x16 output tile (rows [row0,row0+16), cols
// [16*blockIdx.y,+16)); the f32 accumulator lives in 8 VGPRs across all nk
// kernel offsets. Per offset k, the block cooperatively stages W[k]'s column
// slice in LDS as packed bf16 pairs in exact B-fragment order, so each lane's
// fragment is 8 conflict-free stride-1 ds_load_b32s. A-fragment rows come from
// the inverted rulebook gather; inv==-1 rows are zeroed (masked semantics).
// All control flow around WMMA/barriers is block/wave-uniform (EXEC all-ones).
template <int CIN>
__global__ __launch_bounds__(CONV_WAVES * 32) void conv_wmma_kern(
    const float* __restrict__ fin,   // n_in x CIN  (gather source)
    const float* __restrict__ W,     // nk x CIN x cout
    const int*   __restrict__ inv,   // nk x P  (input row or -1)
    const float* __restrict__ bias,  // cout or nullptr
    float* __restrict__ out,         // P x cout
    int P, int cout, int nk)
{
  constexpr int CHUNKS = (CIN + 31) / 32;        // K-steps of 32 (CIN=16 padded)
  constexpr bool H1    = (CIN >= 32);            // upper K half valid?
  __shared__ unsigned int ldsB[CHUNKS * 8 * 32]; // [chunk][t2][lane] bf16 pairs

  const int tid   = threadIdx.x;
  const int lane  = tid & 31;
  const int wv    = tid >> 5;
  const int row0  = (blockIdx.x * CONV_WAVES + wv) * 16;
  const int hf    = lane >> 4;                   // half-wave (ISA K striping)
  const int l16   = lane & 15;
  const int n     = blockIdx.y * 16 + l16;       // output column
  const int myrow = row0 + l16;                  // A-fragment row (M = l16)
  const int koff  = hf * 8;

  v8f c;
  const float bv = bias ? bias[n] : 0.0f;
#pragma unroll
  for (int v = 0; v < 8; ++v) c[v] = bv;

  for (int k = 0; k < nk; ++k) {
    const float* __restrict__ wk = W + (long)k * CIN * cout;
    __syncthreads();
    // cooperative B staging: ldsB[e] = pack(bf16 W[kd][nn], bf16 W[kd+1][nn])
    for (int e = tid; e < CHUNKS * 8 * 32; e += CONV_WAVES * 32) {
      const int cc = e >> 8;
      const int t2 = (e >> 5) & 7;
      const int L  = e & 31;
      const int t  = t2 * 2;
      const int kr = (t & 7) + ((L >> 4) << 3) + ((t & 8) << 1);
      const int kd = cc * 32 + kr;
      const int nn = blockIdx.y * 16 + (L & 15);
      union { __bf16 h[2]; unsigned int u; } pk;
      pk.u = 0;
      if (kd < CIN) {  // compile-time false only for CIN==16 upper half
        pk.h[0] = (__bf16)wk[(long)kd * cout + nn];
        pk.h[1] = (__bf16)wk[(long)(kd + 1) * cout + nn];
      }
      ldsB[e] = pk.u;
    }
    __syncthreads();

    const int idx = (myrow < P) ? inv[k * P + myrow] : -1;
    const float* __restrict__ grow = fin + (long)idx * CIN;
#pragma unroll
    for (int cc = 0; cc < CHUNKS; ++cc) {
      const int k0 = cc * 32 + koff;
      const int k1 = cc * 32 + 16 + koff;
      float a0[8] = {0, 0, 0, 0, 0, 0, 0, 0};
      float a1[8] = {0, 0, 0, 0, 0, 0, 0, 0};
      if (idx >= 0) {
        const float4 u0 = *(const float4*)(grow + k0);
        const float4 u1 = *(const float4*)(grow + k0 + 4);
        a0[0] = u0.x; a0[1] = u0.y; a0[2] = u0.z; a0[3] = u0.w;
        a0[4] = u1.x; a0[5] = u1.y; a0[6] = u1.z; a0[7] = u1.w;
        if (H1) {
          const float4 u2 = *(const float4*)(grow + k1);
          const float4 u3 = *(const float4*)(grow + k1 + 4);
          a1[0] = u2.x; a1[1] = u2.y; a1[2] = u2.z; a1[3] = u2.w;
          a1[4] = u3.x; a1[5] = u3.y; a1[6] = u3.z; a1[7] = u3.w;
        }
      }
      union { v16bf v; unsigned int u[8]; } A, B;
#pragma unroll
      for (int t = 0; t < 8; ++t) {
        A.v[t]     = (__bf16)a0[t];
        A.v[8 + t] = (__bf16)a1[t];
      }
#pragma unroll
      for (int t2 = 0; t2 < 8; ++t2) B.u[t2] = ldsB[cc * 256 + t2 * 32 + lane];
      c = __builtin_amdgcn_wmma_f32_16x16x32_bf16(false, A.v, false, B.v,
                                                  (short)0, c, false, false);
    }
  }
#pragma unroll
  for (int v = 0; v < 8; ++v) {
    const int r = row0 + hf * 8 + v;   // C layout: M = v (+8 for upper half)
    if (r < P) out[(long)r * cout + n] = c[v];
  }
}

// ------------------------------------------------------------- BatchNorm+ReLU
__global__ __launch_bounds__(256) void bn_stats_kern(
    const float* __restrict__ y, long total, int C, float* __restrict__ stats) {
  extern __shared__ float sm[];          // [0,C): sum, [C,2C): sumsq
  for (int i = threadIdx.x; i < 2 * C; i += blockDim.x) sm[i] = 0.0f;
  __syncthreads();
  const int  Cm1    = C - 1;             // C is a power of two
  const long stride = (long)gridDim.x * blockDim.x;
  for (long e = (long)blockIdx.x * blockDim.x + threadIdx.x; e < total; e += stride) {
    const float v = y[e];
    const int   ch = (int)(e & Cm1);
    atomicAdd(&sm[ch], v);
    atomicAdd(&sm[C + ch], v * v);
  }
  __syncthreads();
  for (int i = threadIdx.x; i < 2 * C; i += blockDim.x) atomicAdd(&stats[i], sm[i]);
}

__global__ __launch_bounds__(256) void bn_apply_kern(
    const float* __restrict__ y, float* __restrict__ dst,
    const float* __restrict__ stats, const float* __restrict__ g,
    const float* __restrict__ b, int P, int C) {
  const long total = (long)P * C;
  const int  Cm1   = C - 1;
  const float invP = 1.0f / (float)P;
  const long stride = (long)gridDim.x * blockDim.x;
  for (long e = (long)blockIdx.x * blockDim.x + threadIdx.x; e < total; e += stride) {
    const int   ch  = (int)(e & Cm1);
    const float m   = stats[ch] * invP;
    const float var = stats[C + ch] * invP - m * m;
    const float val = (y[e] - m) * (g[ch] * rsqrtf(var + 1e-5f)) + b[ch];
    dst[e] = fmaxf(val, 0.0f);
  }
}

// --------------------------------------------------------------------- driver
static inline int cdiv_i(int a, int b) { return (a + b - 1) / b; }

static void launch_conv(int cin, dim3 grid, hipStream_t s,
                        const float* fin, const float* W, const int* inv,
                        const float* bias, float* out, int P, int cout, int nk) {
  switch (cin) {
    case 16:
      conv_wmma_kern<16><<<grid, CONV_WAVES * 32, 0, s>>>(fin, W, inv, bias, out, P, cout, nk);
      break;
    case 32:
      conv_wmma_kern<32><<<grid, CONV_WAVES * 32, 0, s>>>(fin, W, inv, bias, out, P, cout, nk);
      break;
    case 64:
      conv_wmma_kern<64><<<grid, CONV_WAVES * 32, 0, s>>>(fin, W, inv, bias, out, P, cout, nk);
      break;
    case 128:
      conv_wmma_kern<128><<<grid, CONV_WAVES * 32, 0, s>>>(fin, W, inv, bias, out, P, cout, nk);
      break;
    default:
      conv_wmma_kern<256><<<grid, CONV_WAVES * 32, 0, s>>>(fin, W, inv, bias, out, P, cout, nk);
      break;
  }
}

extern "C" void kernel_launch(void* const* d_in, const int* in_sizes, int n_in,
                              void* d_out, int out_size, void* d_ws, size_t ws_size,
                              hipStream_t stream) {
  (void)n_in; (void)out_size; (void)ws_size;
  const int CH[5] = {16, 32, 64, 128, 256};
  const float* x = (const float*)d_in[0];
  const int N0 = in_sizes[0] / CH[0];

  int P[4];
  for (int l = 0; l < 4; ++l) P[l] = in_sizes[1 + 14 * l + 1] / 8;  // dI has 8*P elems

  float* out = (float*)d_out;
  hipMemcpyAsync(out, x, (size_t)N0 * CH[0] * sizeof(float),
                 hipMemcpyDeviceToDevice, stream);

  // workspace carve-up (256-byte aligned regions)
  size_t maxPC = 0;
  for (int l = 0; l < 4; ++l) {
    const size_t pc = (size_t)P[l] * CH[l + 1];
    if (pc > maxPC) maxPC = pc;
  }
  char*  ws       = (char*)d_ws;
  size_t invBytes = (((size_t)27 * P[0] * sizeof(int)) + 255) & ~(size_t)255;
  size_t yBytes   = ((maxPC * sizeof(float)) + 255) & ~(size_t)255;
  int*   inv   = (int*)ws;
  float* yA    = (float*)(ws + invBytes);
  float* yB    = (float*)(ws + invBytes + yBytes);
  float* stats = (float*)(ws + invBytes + 2 * yBytes);

  const float* fin = x;
  size_t outOff = (size_t)N0 * CH[0];

  for (int l = 0; l < 4; ++l) {
    const int base = 1 + 14 * l;
    const float*         dW = (const float*)d_in[base + 0];
    const int*           dI = (const int*)d_in[base + 1];
    const int*           dO = (const int*)d_in[base + 2];
    const unsigned char* dM = (const unsigned char*)d_in[base + 3];
    const float*         dG = (const float*)d_in[base + 4];
    const float*         dB = (const float*)d_in[base + 5];
    const float*         bW = (const float*)d_in[base + 7];
    const float*         bC = (const float*)d_in[base + 8];
    const int*           bI = (const int*)d_in[base + 9];
    const int*           bO = (const int*)d_in[base + 10];
    const unsigned char* bM = (const unsigned char*)d_in[base + 11];
    const float*         bG = (const float*)d_in[base + 12];
    const float*         bH = (const float*)d_in[base + 13];
    const int cin = CH[l], cout = CH[l + 1], Pl = P[l];
    const long totPC = (long)Pl * cout;
    const int  redBlocks = (int)((totPC + 255) / 256) < 2048
                           ? (int)((totPC + 255) / 256) : 2048;
    const dim3 convGrid((unsigned)cdiv_i(cdiv_i(Pl, 16), CONV_WAVES),
                        (unsigned)(cout / 16));

    // ---- down conv (8 offsets, no bias) ----
    fill_i32_kern<<<cdiv_i(8 * Pl, 256), 256, 0, stream>>>(inv, 8 * Pl, -1);
    invert_rulebook_kern<<<cdiv_i(8 * Pl, 256), 256, 0, stream>>>(dI, dO, dM, 8 * Pl, Pl, inv);
    launch_conv(cin, convGrid, stream, fin, dW, inv, nullptr, yA, Pl, cout, 8);
    fill_f32_kern<<<cdiv_i(2 * cout, 256), 256, 0, stream>>>(stats, 2 * cout, 0.0f);
    bn_stats_kern<<<redBlocks, 256, 2 * cout * sizeof(float), stream>>>(yA, totPC, cout, stats);
    bn_apply_kern<<<redBlocks, 256, 0, stream>>>(yA, yA, stats, dG, dB, Pl, cout);

    // ---- block conv (27 offsets, bias bC) ----
    fill_i32_kern<<<cdiv_i(27 * Pl, 256), 256, 0, stream>>>(inv, 27 * Pl, -1);
    invert_rulebook_kern<<<cdiv_i(27 * Pl, 256), 256, 0, stream>>>(bI, bO, bM, 27 * Pl, Pl, inv);
    launch_conv(cout, convGrid, stream, yA, bW, inv, bC, yB, Pl, cout, 27);
    fill_f32_kern<<<cdiv_i(2 * cout, 256), 256, 0, stream>>>(stats, 2 * cout, 0.0f);
    bn_stats_kern<<<redBlocks, 256, 2 * cout * sizeof(float), stream>>>(yB, totPC, cout, stats);
    bn_apply_kern<<<redBlocks, 256, 0, stream>>>(yB, out + outOff, stats, bG, bH, Pl, cout);

    fin = out + outOff;        // this level's output feeds the next level
    outOff += (size_t)Pl * cout;
  }
}
